// PathFormer_annotator_88957362635014
// MI455X (gfx1250) — matrix operand order repaired and checked by reading, not compiled, not run on hardware
//
#include <hip/hip_runtime.h>

// Performer FAVOR+ cross-attention forward for MI455X (gfx1250, wave32, WMMA).
// B=8, H=8, Nq=1024, Ns=4096, D=512, DH=64, M=256.

#define B_   8
#define H_   8
#define NQ   1024
#define NS   4096
#define D_   512
#define DH_  64
#define M_   256
#define BH_  64

typedef __bf16 bf16;
typedef __attribute__((ext_vector_type(16))) __bf16 bf16x16;
typedef __attribute__((ext_vector_type(8)))  float  f32x8;

#define WMMA_BF16(a, b, c) \
  __builtin_amdgcn_wmma_f32_16x16x32_bf16(false, (a), false, (b), (short)0, (c), false, false)

union FragU {
  uint4 u[2];
  bf16x16 v;
};
union HalfU {
  uint4 u;
  bf16 h[8];
};

// ---- CDNA5 async global->LDS copy (ASYNCcnt-tracked), 16B per lane ----
// LDS byte address = low 32 bits of the flat shared pointer (LDS aperture is
// 4GB-aligned, ISA 10.2).
__device__ __forceinline__ void async_g2l_b128(void* lds, const void* g) {
  unsigned l = (unsigned)(unsigned long long)lds;
  asm volatile("global_load_async_to_lds_b128 %0, %1, off" :: "v"(l), "v"(g) : "memory");
}
__device__ __forceinline__ void wait_async0() {
  asm volatile("s_wait_asynccnt 0x0" ::: "memory");
}

// ---- fragment loaders (CDNA5 wave32 WMMA VGPR layouts), all 2x ds_load_b128 ----

// A 16x32 bf16 from row-major LDS tile [M][lda] (lda multiple of 8), origin (m0,k0).
__device__ __forceinline__ bf16x16 lds_a_frag(const bf16* s, int lda, int m0, int k0, int lane) {
  const bf16* p = s + (m0 + (lane & 15)) * lda + k0 + ((lane & 16) ? 8 : 0);
  FragU f;
  f.u[0] = *(const uint4*)p;
  f.u[1] = *(const uint4*)(p + 16);
  return f.v;
}

// B 32x16 bf16 from LDS tile stored [N][ldn] ("NK" layout, ldn multiple of 8), origin (k0,n0).
__device__ __forceinline__ bf16x16 lds_b_fragNK(const bf16* s, int ldn, int k0, int n0, int lane) {
  const bf16* p = s + (n0 + (lane & 15)) * ldn + k0 + ((lane & 16) ? 8 : 0);
  FragU f;
  f.u[0] = *(const uint4*)p;
  f.u[1] = *(const uint4*)(p + 16);
  return f.v;
}

// order-preserving f32<->u32 encoding for atomic max over signed floats
__device__ __forceinline__ unsigned f2ord(float f) {
  unsigned u = __float_as_uint(f);
  return (u & 0x80000000u) ? ~u : (u | 0x80000000u);
}
__device__ __forceinline__ float ord2f(unsigned u) {
  unsigned b = (u & 0x80000000u) ? (u & 0x7fffffffu) : ~u;
  return __uint_as_float(b);
}

// stage proj[M_][DH_] f32 -> LDS bf16 [m][d] stride 72 (== NK layout of proj^T)
__device__ __forceinline__ void stage_proj(bf16 (*ps)[72], const float* proj, int tid) {
#pragma unroll 4
  for (int i = 0; i < 16; ++i) {
    int idx = tid + i * 256;            // 4096 float4 chunks
    int m = idx >> 4, d = (idx & 15) * 4;
    float4 v = *(const float4*)(proj + m * DH_ + d);
    ps[m][d] = (bf16)v.x; ps[m][d + 1] = (bf16)v.y;
    ps[m][d + 2] = (bf16)v.z; ps[m][d + 3] = (bf16)v.w;
  }
}

// ==================== Stage 1: QKV projections (f32 in, bf16 out) ====================
__global__ __launch_bounds__(128) void k_gemm_in(const float* __restrict__ A,
                                                 const float* __restrict__ W,
                                                 bf16* __restrict__ out, int nshift) {
  __shared__ __align__(16) bf16 As[64][48];
  __shared__ __align__(16) bf16 BsT[64][40];   // W tile in [N][K] layout
  const int tid = threadIdx.x, lane = tid & 31, wid = tid >> 5;
  const int bm = blockIdx.x * 64;
  const int h = blockIdx.y;
  const int bn = h * 64;
  const int wm = (wid >> 1) * 32, wn = (wid & 1) * 32;
  const int nmask = (1 << nshift) - 1;
  f32x8 acc[2][2] = {};
  for (int kt = 0; kt < D_; kt += 32) {
    __syncthreads();
    float4 va[4], vb[4];
#pragma unroll
    for (int i = 0; i < 4; ++i) {       // batch all global loads first
      int idx = tid + i * 128;
      va[i] = *(const float4*)(A + (size_t)(bm + (idx >> 3)) * D_ + kt + (idx & 7) * 4);
      vb[i] = *(const float4*)(W + (size_t)(kt + (idx >> 4)) * D_ + bn + (idx & 15) * 4);
    }
    if (kt + 32 < D_)  // -> global_prefetch_b8
      __builtin_prefetch(A + (size_t)(bm + (tid >> 1)) * D_ + kt + 32, 0, 1);
#pragma unroll
    for (int i = 0; i < 4; ++i) {       // then convert + stage
      int idx = tid + i * 128;
      int r = idx >> 3, c = (idx & 7) * 4;
      As[r][c] = (bf16)va[i].x; As[r][c + 1] = (bf16)va[i].y;
      As[r][c + 2] = (bf16)va[i].z; As[r][c + 3] = (bf16)va[i].w;
      int rk = idx >> 4, cn = (idx & 15) * 4;
      BsT[cn][rk] = (bf16)vb[i].x; BsT[cn + 1][rk] = (bf16)vb[i].y;
      BsT[cn + 2][rk] = (bf16)vb[i].z; BsT[cn + 3][rk] = (bf16)vb[i].w;
    }
    __syncthreads();
    bf16x16 a0 = lds_a_frag(&As[0][0], 48, wm, 0, lane);
    bf16x16 a1 = lds_a_frag(&As[0][0], 48, wm + 16, 0, lane);
    bf16x16 b0 = lds_b_fragNK(&BsT[0][0], 40, 0, wn, lane);
    bf16x16 b1 = lds_b_fragNK(&BsT[0][0], 40, 0, wn + 16, lane);
    acc[0][0] = WMMA_BF16(a0, b0, acc[0][0]);
    acc[0][1] = WMMA_BF16(a0, b1, acc[0][1]);
    acc[1][0] = WMMA_BF16(a1, b0, acc[1][0]);
    acc[1][1] = WMMA_BF16(a1, b1, acc[1][1]);
  }
#pragma unroll
  for (int i = 0; i < 2; ++i)
#pragma unroll
    for (int j = 0; j < 2; ++j) {
      int dcol = wn + j * 16 + (lane & 15);
      int rbase = bm + wm + i * 16 + ((lane & 16) ? 8 : 0);
#pragma unroll
      for (int r = 0; r < 8; ++r) {
        int grow = rbase + r;
        int bb = grow >> nshift, n = grow & nmask;
        out[((((size_t)(bb * H_ + h)) << nshift) + n) * DH_ + dcol] = (bf16)acc[i][j][r];
      }
    }
}

// ==================== Stage 2: qp feature map (per-row stabilizer) ====================
__global__ __launch_bounds__(256) void k_qp(const bf16* __restrict__ Q,
                                            const float* __restrict__ proj,
                                            bf16* __restrict__ qp) {
  __shared__ __align__(16) bf16 qs[32][72];
  __shared__ __align__(16) bf16 ps[256][72];   // proj^T in [N=m][K=d] layout
  __shared__ float pmax[4][32];
  __shared__ float diag[32], rmax[32];
  const int tid = threadIdx.x, lane = tid & 31, wid = tid >> 5;
  const int bh = blockIdx.y, n0 = blockIdx.x * 32;
  const size_t qbase = ((size_t)bh * NQ + n0) * DH_;
  {
    int r = tid >> 3, c = (tid & 7) * 8;
    async_g2l_b128(&qs[r][c], Q + qbase + (size_t)r * DH_ + c);   // bf16 tile: raw async copy
  }
  stage_proj(ps, proj, tid);
  wait_async0();
  __syncthreads();
  const int mi = (wid >> 2) * 16, ng = (wid & 3) * 4;
  f32x8 acc[4] = {};
#pragma unroll
  for (int ks = 0; ks < DH_; ks += 32) {
    bf16x16 a = lds_a_frag(&qs[0][0], 72, mi, ks, lane);
#pragma unroll
    for (int j = 0; j < 4; ++j) {
      bf16x16 b = lds_b_fragNK(&ps[0][0], 72, ks, (ng + j) * 16, lane);
      acc[j] = WMMA_BF16(a, b, acc[j]);
    }
  }
  const float cnorm = 0.35355339059327379f;  // 64^-0.25
  float mx[8];
#pragma unroll
  for (int r = 0; r < 8; ++r) {
    float m = fmaxf(fmaxf(acc[0][r], acc[1][r]), fmaxf(acc[2][r], acc[3][r]));
#pragma unroll
    for (int d = 1; d < 16; d <<= 1) m = fmaxf(m, __shfl_xor(m, d, 32));
    mx[r] = m * cnorm;
  }
  if ((lane & 15) == 0) {
    int rb = mi + ((lane & 16) ? 8 : 0);
#pragma unroll
    for (int r = 0; r < 8; ++r) pmax[wid & 3][rb + r] = mx[r];
  }
  if (tid < 32) {
    float s = 0.f;
    for (int d = 0; d < DH_; ++d) { float v = (float)qs[tid][d]; s += v * v; }
    diag[tid] = 0.0625f * s;  // 0.5 * 64^-0.5
  }
  __syncthreads();
  if (tid < 32)
    rmax[tid] = fmaxf(fmaxf(pmax[0][tid], pmax[1][tid]), fmaxf(pmax[2][tid], pmax[3][tid]));
  __syncthreads();
  const int rb = mi + ((lane & 16) ? 8 : 0);
#pragma unroll
  for (int j = 0; j < 4; ++j) {
    int col = (ng + j) * 16 + (lane & 15);
#pragma unroll
    for (int r = 0; r < 8; ++r) {
      float v = 0.0625f * (__expf(acc[j][r] * cnorm - diag[rb + r] - rmax[rb + r]) + 1e-4f);
      qp[((size_t)bh * NQ + n0 + rb + r) * M_ + col] = (bf16)v;
    }
  }
}

// ==================== Stage 3: global stabilizer for keys ====================
__global__ void k_init(unsigned* stab) { stab[threadIdx.x] = 0u; }

__global__ __launch_bounds__(256) void k_kstab(const bf16* __restrict__ K,
                                               const float* __restrict__ proj,
                                               unsigned* __restrict__ stab) {
  __shared__ __align__(16) bf16 ks_[32][72];
  __shared__ __align__(16) bf16 ps[256][72];
  __shared__ float red[8];
  const int tid = threadIdx.x, lane = tid & 31, wid = tid >> 5;
  const int bh = blockIdx.y, s0 = blockIdx.x * 32;
  const size_t kbase = ((size_t)bh * NS + s0) * DH_;
  {
    int r = tid >> 3, c = (tid & 7) * 8;
    async_g2l_b128(&ks_[r][c], K + kbase + (size_t)r * DH_ + c);
  }
  stage_proj(ps, proj, tid);
  wait_async0();
  __syncthreads();
  const int mi = (wid >> 2) * 16, ng = (wid & 3) * 4;
  f32x8 acc[4] = {};
#pragma unroll
  for (int ks = 0; ks < DH_; ks += 32) {
    bf16x16 a = lds_a_frag(&ks_[0][0], 72, mi, ks, lane);
#pragma unroll
    for (int j = 0; j < 4; ++j) {
      bf16x16 b = lds_b_fragNK(&ps[0][0], 72, ks, (ng + j) * 16, lane);
      acc[j] = WMMA_BF16(a, b, acc[j]);
    }
  }
  float m = -3.0e38f;
#pragma unroll
  for (int j = 0; j < 4; ++j)
#pragma unroll
    for (int r = 0; r < 8; ++r) m = fmaxf(m, acc[j][r]);
#pragma unroll
  for (int d = 1; d < 32; d <<= 1) m = fmaxf(m, __shfl_xor(m, d, 32));
  if (lane == 0) red[wid] = m;
  __syncthreads();
  if (tid == 0) {
    float v = red[0];
#pragma unroll
    for (int w = 1; w < 8; ++w) v = fmaxf(v, red[w]);
    atomicMax(&stab[bh], f2ord(v * 0.35355339059327379f));
  }
}

// ==================== Stage 4: fused kp + ctx + k_cumsum (persistent per b,h) ====
__global__ __launch_bounds__(256) void k_kvctx(const bf16* __restrict__ K,
                                               const bf16* __restrict__ V,
                                               const float* __restrict__ proj,
                                               const unsigned* __restrict__ stab_o,
                                               float* __restrict__ kcum_out,
                                               bf16* __restrict__ ctx_out) {
  __shared__ __align__(16) bf16 Bs[256][72];   // proj^T staged once, [m][d] NK layout
  __shared__ __align__(16) bf16 Ks[32][72];
  __shared__ __align__(16) bf16 VsT[64][40];   // V chunk, [d][s] NK layout
  __shared__ __align__(16) bf16 kpT[256][32];  // kp^T, [m][s] row-major A layout
  __shared__ float kcum[256];
  __shared__ float diag[32];
  const int tid = threadIdx.x, lane = tid & 31, wid = tid >> 5;
  const int bh = blockIdx.x;
  const float stab = ord2f(stab_o[bh]);
  stage_proj(Bs, proj, tid);
  kcum[tid] = 0.f;
  const size_t base = (size_t)bh * NS * DH_;
  const int dd_mi = (wid >> 2) * 16, dd_ng = (wid & 3) * 4;
  const int cm = wid * 32;  // this wave's ctx M-range [cm, cm+32)
  const int lr = tid >> 3, lc = (tid & 7) * 8;
  f32x8 ctx[2][4] = {};
  for (int sc = 0; sc < NS; sc += 32) {
    __syncthreads();
    async_g2l_b128(&Ks[lr][lc], K + base + (size_t)(sc + lr) * DH_ + lc);  // K: raw async copy
    {
      HalfU hv;  // V needs a transpose in LDS -> manual
      hv.u = *(const uint4*)(V + base + (size_t)(sc + lr) * DH_ + lc);
#pragma unroll
      for (int k = 0; k < 8; ++k) VsT[lc + k][lr] = hv.h[k];
    }
    if (sc + 32 < NS)
      __builtin_prefetch(V + base + (size_t)(sc + 32 + lr) * DH_, 0, 1);
    wait_async0();
    __syncthreads();
    if (tid < 32) {
      float s = 0.f;
      for (int d = 0; d < DH_; ++d) { float v = (float)Ks[tid][d]; s += v * v; }
      diag[tid] = 0.0625f * s;
    }
    f32x8 t[4] = {};
#pragma unroll
    for (int ksi = 0; ksi < DH_; ksi += 32) {
      bf16x16 a = lds_a_frag(&Ks[0][0], 72, dd_mi, ksi, lane);
#pragma unroll
      for (int j = 0; j < 4; ++j) {
        bf16x16 b = lds_b_fragNK(&Bs[0][0], 72, ksi, (dd_ng + j) * 16, lane);
        t[j] = WMMA_BF16(a, b, t[j]);
      }
    }
    __syncthreads();  // diag ready
#pragma unroll
    for (int j = 0; j < 4; ++j) {
      int col = (dd_ng + j) * 16 + (lane & 15);            // m index
      int rb = dd_mi + ((lane & 16) ? 8 : 0);              // s index base
#pragma unroll
      for (int r = 0; r < 8; ++r) {
        float v = t[j][r] * 0.35355339059327379f - diag[rb + r] - stab;
        kpT[col][rb + r] = (bf16)(0.0625f * (__expf(v) + 1e-4f));
      }
    }
    __syncthreads();  // kp ready
    {
      float s = 0.f;
#pragma unroll 8
      for (int r = 0; r < 32; ++r) s += (float)kpT[tid][r];
      kcum[tid] += s;
    }
#pragma unroll
    for (int i = 0; i < 2; ++i) {
      bf16x16 a = lds_a_frag(&kpT[0][0], 32, cm + i * 16, 0, lane);  // A = kp^T
#pragma unroll
      for (int j = 0; j < 4; ++j) {
        bf16x16 b = lds_b_fragNK(&VsT[0][0], 40, 0, j * 16, lane);
        ctx[i][j] = WMMA_BF16(a, b, ctx[i][j]);
      }
    }
  }
  __syncthreads();
  kcum_out[bh * M_ + tid] = kcum[tid];
#pragma unroll
  for (int i = 0; i < 2; ++i)
#pragma unroll
    for (int j = 0; j < 4; ++j) {
      int m = cm + i * 16 + ((lane & 16) ? 8 : 0);
      int dcol = j * 16 + (lane & 15);
#pragma unroll
      for (int r = 0; r < 8; ++r)
        ctx_out[((size_t)bh * M_ + m + r) * DH_ + dcol] = (bf16)ctx[i][j][r];
    }
}

// ==================== Stage 5: out = (qp @ ctx) * d_inv -> [b,n,512] bf16 ==========
__global__ __launch_bounds__(256) void k_attn(const bf16* __restrict__ qp,
                                              const bf16* __restrict__ ctx,
                                              const float* __restrict__ kcum,
                                              bf16* __restrict__ attn) {
  __shared__ __align__(16) bf16 qs[32][264];
  __shared__ __align__(16) bf16 csT[64][264];  // ctx in [N=d][K=m] layout
  __shared__ float kc[256];
  __shared__ float dinv[32];
  const int tid = threadIdx.x, lane = tid & 31, wid = tid >> 5;
  const int bh = blockIdx.y, n0 = blockIdx.x * 32;
#pragma unroll
  for (int i = 0; i < 4; ++i) {
    int idx = tid + i * 256;
    int r = idx >> 5, c = (idx & 31) * 8;
    async_g2l_b128(&qs[r][c], qp + ((size_t)bh * NQ + n0 + r) * M_ + c);  // raw async copy
  }
#pragma unroll
  for (int i = 0; i < 8; ++i) {
    int idx = tid + i * 256;
    int r = idx >> 3, c = (idx & 7) * 8;   // r = m, c = d offset
    HalfU hv;
    hv.u = *(const uint4*)(ctx + ((size_t)bh * M_ + r) * DH_ + c);
#pragma unroll
    for (int k = 0; k < 8; ++k) csT[c + k][r] = hv.h[k];
  }
  kc[tid] = kcum[bh * M_ + tid];
  wait_async0();
  __syncthreads();
  if (tid < 32) {
    float s = 0.f;
    for (int m = 0; m < M_; ++m) s += (float)qs[tid][m] * kc[m];
    dinv[tid] = 1.0f / s;
  }
  __syncthreads();
  const int mi = (wid >> 2) * 16, nj = (wid & 3) * 16;
  f32x8 acc = {};
#pragma unroll
  for (int ks = 0; ks < M_; ks += 32) {
    bf16x16 a = lds_a_frag(&qs[0][0], 264, mi, ks, lane);
    bf16x16 b = lds_b_fragNK(&csT[0][0], 264, ks, nj, lane);
    acc = WMMA_BF16(a, b, acc);
  }
  const int bb = bh >> 3, h = bh & 7;
  const int dcol = nj + (lane & 15);
  const int rb = mi + ((lane & 16) ? 8 : 0);
#pragma unroll
  for (int r = 0; r < 8; ++r) {
    int n = n0 + rb + r;
    attn[((size_t)bb * NQ + n) * D_ + h * DH_ + dcol] = (bf16)(acc[r] * dinv[rb + r]);
  }
}

// ==================== Stage 6: final projection (bf16 A, + bias, f32 out) =========
__global__ __launch_bounds__(128) void k_gemm_out(const bf16* __restrict__ A,
                                                  const float* __restrict__ W,
                                                  const float* __restrict__ bias,
                                                  float* __restrict__ out) {
  __shared__ __align__(16) bf16 As[64][48];
  __shared__ __align__(16) bf16 BsT[64][40];
  const int tid = threadIdx.x, lane = tid & 31, wid = tid >> 5;
  const int bm = blockIdx.x * 64, bn = blockIdx.y * 64;
  const int wm = (wid >> 1) * 32, wn = (wid & 1) * 32;
  f32x8 acc[2][2] = {};
  for (int kt = 0; kt < D_; kt += 32) {
    __syncthreads();
#pragma unroll
    for (int i = 0; i < 2; ++i) {
      int idx = tid + i * 128;
      int r = idx >> 2, c = (idx & 3) * 8;
      async_g2l_b128(&As[r][c], A + (size_t)(bm + r) * D_ + kt + c);  // bf16: async copy
    }
    float4 vb[4];
#pragma unroll
    for (int i = 0; i < 4; ++i) {
      int idx = tid + i * 128;
      vb[i] = *(const float4*)(W + (size_t)(kt + (idx >> 4)) * D_ + bn + (idx & 15) * 4);
    }
    if (kt + 32 < D_)
      __builtin_prefetch(A + (size_t)(bm + (tid >> 1)) * D_ + kt + 32, 0, 1);
#pragma unroll
    for (int i = 0; i < 4; ++i) {
      int idx = tid + i * 128;
      int r = idx >> 4, c = (idx & 15) * 4;
      BsT[c][r] = (bf16)vb[i].x; BsT[c + 1][r] = (bf16)vb[i].y;
      BsT[c + 2][r] = (bf16)vb[i].z; BsT[c + 3][r] = (bf16)vb[i].w;
    }
    wait_async0();
    __syncthreads();
    bf16x16 a0 = lds_a_frag(&As[0][0], 48, wm, 0, lane);
    bf16x16 a1 = lds_a_frag(&As[0][0], 48, wm + 16, 0, lane);
    bf16x16 b0 = lds_b_fragNK(&BsT[0][0], 40, 0, wn, lane);
    bf16x16 b1 = lds_b_fragNK(&BsT[0][0], 40, 0, wn + 16, lane);
    acc[0][0] = WMMA_BF16(a0, b0, acc[0][0]);
    acc[0][1] = WMMA_BF16(a0, b1, acc[0][1]);
    acc[1][0] = WMMA_BF16(a1, b0, acc[1][0]);
    acc[1][1] = WMMA_BF16(a1, b1, acc[1][1]);
  }
#pragma unroll
  for (int i = 0; i < 2; ++i)
#pragma unroll
    for (int j = 0; j < 2; ++j) {
      int col = bn + wn + j * 16 + (lane & 15);
      int rb = bm + wm + i * 16 + ((lane & 16) ? 8 : 0);
      float bv = bias[col];
#pragma unroll
      for (int r = 0; r < 8; ++r)
        out[(size_t)(rb + r) * D_ + col] = acc[i][j][r] + bv;
    }
}

// ==================== host launcher ====================
extern "C" void kernel_launch(void* const* d_in, const int* in_sizes, int n_in,
                              void* d_out, int out_size, void* d_ws, size_t ws_size,
                              hipStream_t stream) {
  (void)in_sizes; (void)n_in; (void)out_size; (void)ws_size;
  const float* x    = (const float*)d_in[0];
  const float* cx   = (const float*)d_in[1];
  const float* Wq   = (const float*)d_in[2];
  const float* Wk   = (const float*)d_in[3];
  const float* Wv   = (const float*)d_in[4];
  const float* Wo   = (const float*)d_in[5];
  const float* bo   = (const float*)d_in[6];
  const float* proj = (const float*)d_in[7];
  float* out = (float*)d_out;

  char* p = (char*)d_ws;
  auto take = [&](size_t bytes) {
    char* r = p;
    p += (bytes + 255) & ~(size_t)255;
    return (void*)r;
  };
  bf16* Qb       = (bf16*)take((size_t)B_ * H_ * NQ * DH_ * 2);
  bf16* Kb       = (bf16*)take((size_t)B_ * H_ * NS * DH_ * 2);
  bf16* Vb       = (bf16*)take((size_t)B_ * H_ * NS * DH_ * 2);
  bf16* QPb      = (bf16*)take((size_t)B_ * H_ * NQ * M_ * 2);
  unsigned* stab = (unsigned*)take((size_t)BH_ * 4);
  float* kcum    = (float*)take((size_t)BH_ * M_ * 4);
  bf16* ctxb     = (bf16*)take((size_t)BH_ * M_ * DH_ * 2);
  bf16* attn     = (bf16*)take((size_t)B_ * NQ * D_ * 2);

  k_gemm_in<<<dim3(B_ * NQ / 64, H_), 128, 0, stream>>>(x,  Wq, Qb, 10);
  k_gemm_in<<<dim3(B_ * NS / 64, H_), 128, 0, stream>>>(cx, Wk, Kb, 12);
  k_gemm_in<<<dim3(B_ * NS / 64, H_), 128, 0, stream>>>(cx, Wv, Vb, 12);
  k_init<<<1, BH_, 0, stream>>>(stab);
  k_qp<<<dim3(NQ / 32, BH_), 256, 0, stream>>>(Qb, proj, QPb);
  k_kstab<<<dim3(NS / 32, BH_), 256, 0, stream>>>(Kb, proj, stab);
  k_kvctx<<<dim3(BH_), 256, 0, stream>>>(Kb, Vb, proj, stab, kcum, ctxb);
  k_attn<<<dim3(NQ / 32, BH_), 256, 0, stream>>>(QPb, ctxb, kcum, attn);
  k_gemm_out<<<dim3(B_ * NQ / 64, H_), 128, 0, stream>>>(attn, Wo, bo, out);
}
